// ModernNCA_75488345194600
// MI455X (gfx1250) — matrix operand order repaired and checked by reading, not compiled, not run on hardware
//
#include <hip/hip_runtime.h>
#include <hip/hip_bf16.h>
#include <math.h>

// ---------------- problem constants (match reference) ----------------
#define BSZ   1024      // batch rows
#define MCAND 200000    // candidates
#define DIN   256
#define DIMH  128
#define DOUT  10
#define EPSV  1e-7f

#define NPART 50                          // candidate partitions
#define NTILES (MCAND / 16)               // 12500 candidate tiles
#define TILES_PER_PART (NTILES / NPART)   // 250

typedef float  v2f   __attribute__((ext_vector_type(2)));
typedef float  v8f   __attribute__((ext_vector_type(8)));
typedef __bf16 v16bf __attribute__((ext_vector_type(16)));

// ---------------------------------------------------------------------
// Encoder GEMM (f32 WMMA 16x16x4) + split-bf16 pack + row sq-norms.
//   H[row,n] = sum_k X[row,k] * W[n,k] + bias[n]
// One 16-row tile per workgroup; 8 waves each own 16 of the 128 dims.
// Each f32 value v is split v = hi + lo (bf16 each) and stored to Phi/Plo.
//   a_layout=1 (candidates / WMMA-A side): same plain row-major order, K
//     in natural order, so a lane's 16-elem bf16 A (K 16h..16h+15 of a 32-K chunk)
//     fragment is one contiguous 32B load.
//   a_layout=0 (batch / WMMA-B side): plain row-major (B read the same way as A:
//     lane half <-> 16 consecutive K at column n).
// Row squared norms reduced via LDS ds_add_f32.
// ---------------------------------------------------------------------
__global__ __launch_bounds__(256) void encode_pack_kernel(
    const float* __restrict__ X, const float* __restrict__ W,
    const float* __restrict__ bias,
    __bf16* __restrict__ Phi, __bf16* __restrict__ Plo,
    float* __restrict__ SQ, int a_layout)
{
  const int row0 = blockIdx.x * 16;
  const int wave = threadIdx.x >> 5;
  const int lane = threadIdx.x & 31;
  const int m    = lane & 15;
  const int hi   = lane >> 4;          // 0: K+{0,1}, 1: K+{2,3}
  const int n0   = wave * 16;

  __shared__ float rsq[16];
  if (threadIdx.x < 16) rsq[threadIdx.x] = 0.f;
  __syncthreads();

  const float* __restrict__ xrow = X + (size_t)(row0 + m) * DIN;
  const float* __restrict__ wrow = W + (size_t)(n0 + m) * DIN;   // B[k][n] = W[n][k]

  v8f acc = {0.f,0.f,0.f,0.f,0.f,0.f,0.f,0.f};
  #pragma unroll 4
  for (int k0 = 0; k0 < DIN; k0 += 4) {
    const int ka = k0 + hi * 2;
    v2f a = { xrow[ka], xrow[ka + 1] };
    v2f b = { wrow[ka], wrow[ka + 1] };
    acc = __builtin_amdgcn_wmma_f32_16x16x4_f32(false, a, false, b,
                                                (short)0, acc, false, false);
  }

  // destination column for this lane's output feature n
  int dstcol;
  {
    const int n = n0 + m;
    if (a_layout) {
      const int c4 = n >> 5, j = n & 31, g = j >> 3;
      dstcol = c4 * 32 + j;
    } else {
      dstcol = n;
    }
  }

  const float bv = bias[n0 + m];
  #pragma unroll
  for (int r = 0; r < 8; ++r) {
    const int   lr = hi * 8 + r;               // local row 0..15
    const float v  = acc[r] + bv;
    atomicAdd(&rsq[lr], v * v);                // ds_add_f32
    const __bf16 vh = (__bf16)v;
    const __bf16 vl = (__bf16)(v - (float)vh);
    const size_t dst = (size_t)(row0 + lr) * DIMH + dstcol;
    Phi[dst] = vh;
    Plo[dst] = vl;
  }

  __syncthreads();
  if (threadIdx.x < 16) SQ[row0 + threadIdx.x] = rsq[threadIdx.x];
}

// ---------------------------------------------------------------------
// Fused distance -> exp -> class accumulation, split-bf16 WMMA.
// grid = (16 batch-blocks of 64 rows, NPART candidate partitions), 256 thr.
// Per wave / candidate tile: A = 16 candidates x 128 K (bf16 hi/lo),
// 4 batch sub-tiles q. dot = Ahi*Bhi + Ahi*Blo + Alo*Bhi via
// v_wmma_f32_16x16x32_bf16 (12 WMMA per 16x16x128 tile instead of 32 f32).
// D[q] lane layout: (cand r + 8*hi) x (batch col q*16 + m).
// exp(-dist) accumulated into LDS red[64][10] via ds_add_f32; the softmax
// denominator is the class-sum (shift-free softmax is exact here: dist~16).
// ---------------------------------------------------------------------
__global__ __launch_bounds__(256) void nca_partial_kernel(
    const __bf16* __restrict__ Bhi, const __bf16* __restrict__ Blo,   // batch h, row-major
    const __bf16* __restrict__ Ahi, const __bf16* __restrict__ Alo,   // candidates, row-major
    const float* __restrict__ hsq, const float* __restrict__ csq,
    const int*   __restrict__ cy,  float* __restrict__ partials)
{
  const int bblk = blockIdx.x;     // batch rows [bblk*64, bblk*64+64)
  const int part = blockIdx.y;
  const int wave = threadIdx.x >> 5;
  const int lane = threadIdx.x & 31;
  const int m    = lane & 15;
  const int hi   = lane >> 4;

  __shared__ float red[64][DOUT];
  for (int i = threadIdx.x; i < 64 * DOUT; i += 256) ((float*)red)[i] = 0.f;
  __syncthreads();

  // batch-side fragment pointers / norms for this lane's 4 column sub-tiles
  const __bf16* __restrict__ bh_q[4];
  const __bf16* __restrict__ bl_q[4];
  float hsq_l[4];
  #pragma unroll
  for (int q = 0; q < 4; ++q) {
    const int brow = bblk * 64 + q * 16 + m;
    bh_q[q]  = Bhi + (size_t)brow * DIMH;
    bl_q[q]  = Blo + (size_t)brow * DIMH;
    hsq_l[q] = hsq[brow];
  }

  const int t0   = part * TILES_PER_PART;
  const int tEnd = t0 + TILES_PER_PART;

  for (int t = t0 + wave; t < tEnd; t += 8) {
    const int cand0 = t * 16;
    const __bf16* __restrict__ arh = Ahi + (size_t)(cand0 + m) * DIMH;
    const __bf16* __restrict__ arl = Alo + (size_t)(cand0 + m) * DIMH;
    // prefetch this lane's candidate rows for the next tile we will process
    __builtin_prefetch(arh + (size_t)8 * 16 * DIMH, 0, 1);
    __builtin_prefetch(arl + (size_t)8 * 16 * DIMH, 0, 1);

    v8f d0 = {0.f,0.f,0.f,0.f,0.f,0.f,0.f,0.f};
    v8f d1 = d0, d2 = d0, d3 = d0;

    #pragma unroll
    for (int c4 = 0; c4 < 4; ++c4) {
      const int off = c4 * 32 + hi * 16;   // element offset within a row
      const v16bf ah = *(const v16bf*)(arh + off);
      const v16bf al = *(const v16bf*)(arl + off);
      {
        const v16bf bh = *(const v16bf*)(bh_q[0] + off);
        const v16bf bl = *(const v16bf*)(bl_q[0] + off);
        d0 = __builtin_amdgcn_wmma_f32_16x16x32_bf16(false, ah, false, bh, (short)0, d0, false, false);
        d0 = __builtin_amdgcn_wmma_f32_16x16x32_bf16(false, ah, false, bl, (short)0, d0, false, false);
        d0 = __builtin_amdgcn_wmma_f32_16x16x32_bf16(false, al, false, bh, (short)0, d0, false, false);
      }
      {
        const v16bf bh = *(const v16bf*)(bh_q[1] + off);
        const v16bf bl = *(const v16bf*)(bl_q[1] + off);
        d1 = __builtin_amdgcn_wmma_f32_16x16x32_bf16(false, ah, false, bh, (short)0, d1, false, false);
        d1 = __builtin_amdgcn_wmma_f32_16x16x32_bf16(false, ah, false, bl, (short)0, d1, false, false);
        d1 = __builtin_amdgcn_wmma_f32_16x16x32_bf16(false, al, false, bh, (short)0, d1, false, false);
      }
      {
        const v16bf bh = *(const v16bf*)(bh_q[2] + off);
        const v16bf bl = *(const v16bf*)(bl_q[2] + off);
        d2 = __builtin_amdgcn_wmma_f32_16x16x32_bf16(false, ah, false, bh, (short)0, d2, false, false);
        d2 = __builtin_amdgcn_wmma_f32_16x16x32_bf16(false, ah, false, bl, (short)0, d2, false, false);
        d2 = __builtin_amdgcn_wmma_f32_16x16x32_bf16(false, al, false, bh, (short)0, d2, false, false);
      }
      {
        const v16bf bh = *(const v16bf*)(bh_q[3] + off);
        const v16bf bl = *(const v16bf*)(bl_q[3] + off);
        d3 = __builtin_amdgcn_wmma_f32_16x16x32_bf16(false, ah, false, bh, (short)0, d3, false, false);
        d3 = __builtin_amdgcn_wmma_f32_16x16x32_bf16(false, ah, false, bl, (short)0, d3, false, false);
        d3 = __builtin_amdgcn_wmma_f32_16x16x32_bf16(false, al, false, bh, (short)0, d3, false, false);
      }
    }

    const int cbase = cand0 + hi * 8;
    #pragma unroll
    for (int r = 0; r < 8; ++r) {
      const float cq  = csq[cbase + r];
      const int   cls = cy[cbase + r];
      #pragma unroll
      for (int q = 0; q < 4; ++q) {
        const float dot = (q == 0) ? d0[r] : (q == 1) ? d1[r] : (q == 2) ? d2[r] : d3[r];
        float sq = hsq_l[q] + cq - 2.0f * dot;
        sq = fmaxf(sq, 0.0f);
        const float p = __expf(-__fsqrt_rn(sq));   // T = 1.0
        atomicAdd(&red[q * 16 + m][cls], p);       // ds_add_f32, no return
      }
    }
  }

  __syncthreads();
  // write partial per-(row,class) sums: partials[part][row][class]
  for (int i = threadIdx.x; i < 64 * DOUT; i += 256) {
    const int row = bblk * 64 + i / DOUT;
    partials[((size_t)part * BSZ + row) * DOUT + (i % DOUT)] = ((float*)red)[i];
  }
}

// ---------------------------------------------------------------------
// Finalize: sum partials over partitions, normalize, log(p + eps).
// One wave per batch row; lanes 0..9 own one class each.
// ---------------------------------------------------------------------
__global__ __launch_bounds__(32) void finalize_kernel(
    const float* __restrict__ partials, float* __restrict__ out)
{
  const int row  = blockIdx.x;
  const int lane = threadIdx.x;
  float a = 0.f;
  if (lane < DOUT) {
    for (int p = 0; p < NPART; ++p)
      a += partials[((size_t)p * BSZ + row) * DOUT + lane];
  }
  float s = a;
  #pragma unroll
  for (int off = 16; off; off >>= 1) s += __shfl_xor(s, off, 32);
  if (lane < DOUT) out[(size_t)row * DOUT + lane] = logf(a / s + EPSV);
}

// ---------------------------------------------------------------------
extern "C" void kernel_launch(void* const* d_in, const int* in_sizes, int n_in,
                              void* d_out, int out_size, void* d_ws, size_t ws_size,
                              hipStream_t stream) {
  const float* x  = (const float*)d_in[0];   // [1024, 256]
  // d_in[1] = y (unused on eval path)
  const float* cx = (const float*)d_in[2];   // [200000, 256]
  const int*   cy = (const int*)d_in[3];     // [200000]
  const float* W  = (const float*)d_in[4];   // [128, 256]
  const float* be = (const float*)d_in[5];   // [128]
  // d_in[6] = is_train (0 => eval path)
  float* out = (float*)d_out;                // [1024, 10]

  // workspace layout (all chunks 32B-aligned)
  char* ws = (char*)d_ws;
  __bf16* hB_hi  = (__bf16*)ws;                 ws += (size_t)BSZ   * DIMH * 2;  // 256KB
  __bf16* hB_lo  = (__bf16*)ws;                 ws += (size_t)BSZ   * DIMH * 2;  // 256KB
  __bf16* hcA_hi = (__bf16*)ws;                 ws += (size_t)MCAND * DIMH * 2;  // 51.2MB
  __bf16* hcA_lo = (__bf16*)ws;                 ws += (size_t)MCAND * DIMH * 2;  // 51.2MB
  float*  hsq    = (float*)ws;                  ws += (size_t)BSZ * 4;
  float*  csq    = (float*)ws;                  ws += (size_t)MCAND * 4;
  float*  partials = (float*)ws;                // NPART*1024*10 floats (~2MB)

  encode_pack_kernel<<<BSZ / 16,   256, 0, stream>>>(x,  W, be, hB_hi,  hB_lo,  hsq, 0);
  encode_pack_kernel<<<MCAND / 16, 256, 0, stream>>>(cx, W, be, hcA_hi, hcA_lo, csq, 1);
  nca_partial_kernel<<<dim3(BSZ / 64, NPART), 256, 0, stream>>>(
      hB_hi, hB_lo, hcA_hi, hcA_lo, hsq, csq, cy, partials);
  finalize_kernel<<<BSZ, 32, 0, stream>>>(partials, out);
}
